// BertSelfAttention_77257871720475
// MI455X (gfx1250) — matrix operand order repaired and checked by reading, not compiled, not run on hardware
//
#include <hip/hip_runtime.h>
#include <hip/hip_bf16.h>

// ---------------------------------------------------------------------------
// BERT self-attention with relative_key_query position bias, MI455X (gfx1250).
// All matrix math on v_wmma_f32_16x16x32_bf16 (f32 accumulate).
// B=4, S=1024, H=768, 12 heads x 64 dim, dist_emb rows = 2047.
// ---------------------------------------------------------------------------

#define BATCH 4
#define SEQ   1024
#define HID   768
#define NH    12
#define HD    64
#define NTOK  (BATCH * SEQ)        // 4096
#define EROWS (2 * 1024 - 1)       // 2047

typedef __bf16 bf16;
typedef __attribute__((ext_vector_type(16))) __bf16 v16bf;
typedef __attribute__((ext_vector_type(8)))  float  v8f;
typedef __attribute__((ext_vector_type(4)))  float  f4;

static_assert(sizeof(v16bf) == 32, "v16bf must be 32 bytes");

struct P2 { f4 a, b; };

// 16 bf16 from two independent 16-byte chunks (A-operand, same k order as B:
// lane holds row M=lane%16, K runs [16*(lane/16),+8) and [16*(lane/16)+8,+8)).
__device__ inline v16bf ld_ab(const bf16* p0, const bf16* p1) {
  P2 t;
  t.a = *reinterpret_cast<const f4*>(p0);
  t.b = *reinterpret_cast<const f4*>(p1);
  return __builtin_bit_cast(v16bf, t);
}
// 16 contiguous bf16 (B-operand layout: lane holds col N=lane%16,
// K run of 16 contiguous values at 16*(lane/16)).
__device__ inline v16bf ld_b16(const bf16* p) {
  P2 t;
  t.a = *reinterpret_cast<const f4*>(p);
  t.b = *reinterpret_cast<const f4*>(p + 8);
  return __builtin_bit_cast(v16bf, t);
}

__device__ inline v8f wmma_bf16(v16bf a, v16bf b, v8f c) {
  return __builtin_amdgcn_wmma_f32_16x16x32_bf16(
      false, a, false, b, (short)0, c, false, false);
}

// ---------------------------------------------------------------------------
// Workspace layout (bytes, all 128-aligned). Total ~27.6 MB.
// ---------------------------------------------------------------------------
#define OFF_HB   ((size_t)0)                    // hidden bf16     [4096][768]
#define OFF_WTQ  ((size_t)6291456)              // Wq^T bf16       [768][768]
#define OFF_WTK  ((size_t)7471104)
#define OFF_WTV  ((size_t)8650752)
#define OFF_EB   ((size_t)9830400)              // dist_emb bf16   [2047][64]
#define OFF_Q    ((size_t)10092416)             // Q bf16  [B][h][s][d]
#define OFF_K    ((size_t)16383872)             // K bf16  [B][h][s][d]
#define OFF_VT   ((size_t)22675328)             // V^T bf16 [B][h][d][s]

// ---------------------------------------------------------------------------
// f32 -> bf16 converters
// ---------------------------------------------------------------------------
__global__ void k_cvt(const float* __restrict__ src, bf16* __restrict__ dst, int n) {
  for (int i = blockIdx.x * blockDim.x + threadIdx.x; i < n;
       i += gridDim.x * blockDim.x)
    dst[i] = (bf16)src[i];
}

// Wt[o][i] = W[i][o]  (so GEMM B-operands are contiguous per lane)
__global__ void k_cvt_wt(const float* __restrict__ W, bf16* __restrict__ Wt) {
  int i = blockIdx.x * blockDim.x + threadIdx.x;
  if (i < HID * HID) {
    int o = i / HID, k = i - o * HID;
    Wt[i] = (bf16)W[k * HID + o];
  }
}

// ---------------------------------------------------------------------------
// QKV projection: one wave -> one 16x64 output tile (4 accumulators).
// A (token tile) is loaded once per k-step and reused across 4 WMMAs.
// K-loop over 768: 24 steps x 4 WMMAs = 96 WMMAs per wave.
// Q,K stored [b][h][s][d]; V stored transposed [b][h][d][s] for the PV pass.
// ---------------------------------------------------------------------------
__global__ void __launch_bounds__(128)
k_qkv(const bf16* __restrict__ hb,
      const bf16* __restrict__ wtq, const bf16* __restrict__ wtk,
      const bf16* __restrict__ wtv,
      const float* __restrict__ bq, const float* __restrict__ bk,
      const float* __restrict__ bv,
      bf16* __restrict__ qo, bf16* __restrict__ ko, bf16* __restrict__ vto) {
  const int lane = threadIdx.x & 31, wid = threadIdx.x >> 5;
  const int lh = lane >> 4, ll = lane & 15;
  const int MT = NTOK / 16, NT = HID / 64;          // 256, 12
  int wg  = blockIdx.x * 4 + wid;                   // 0 .. 3*MT*NT-1
  int mat = wg / (MT * NT);
  int rem = wg - mat * (MT * NT);
  int mt  = rem / NT, nt = rem - mt * NT;           // nt: 64-wide column tile

  const bf16* wt   = (mat == 0) ? wtq : (mat == 1) ? wtk : wtv;
  const float* bia = (mat == 0) ? bq  : (mat == 1) ? bk  : bv;

  const bf16* arow = hb + (size_t)(mt * 16 + ll) * HID;        // token row (A)
  const bf16* brow0 = wt + (size_t)(nt * 64 +  0 + ll) * HID;  // feature rows (B)
  const bf16* brow1 = wt + (size_t)(nt * 64 + 16 + ll) * HID;
  const bf16* brow2 = wt + (size_t)(nt * 64 + 32 + ll) * HID;
  const bf16* brow3 = wt + (size_t)(nt * 64 + 48 + ll) * HID;

  v8f acc0 = {}, acc1 = {}, acc2 = {}, acc3 = {};
  for (int k0 = 0; k0 < HID; k0 += 32) {
    v16bf a = ld_ab(arow + k0 + 16 * lh, arow + k0 + 16 * lh + 8);
    acc0 = wmma_bf16(a, ld_b16(brow0 + k0 + 16 * lh), acc0);
    acc1 = wmma_bf16(a, ld_b16(brow1 + k0 + 16 * lh), acc1);
    acc2 = wmma_bf16(a, ld_b16(brow2 + k0 + 16 * lh), acc2);
    acc3 = wmma_bf16(a, ld_b16(brow3 + k0 + 16 * lh), acc3);
  }

  // n0 is 64-aligned, so the whole 16x64 tile belongs to one head.
  int h = (nt * 64) >> 6;
  float bias0 = bia[nt * 64 +  0 + ll];
  float bias1 = bia[nt * 64 + 16 + ll];
  float bias2 = bia[nt * 64 + 32 + ll];
  float bias3 = bia[nt * 64 + 48 + ll];
#pragma unroll
  for (int v = 0; v < 8; ++v) {
    int tok = mt * 16 + v + 8 * lh;                 // C layout: M = v + 8*(lane/16)
    int b_  = tok >> 10, s = tok & (SEQ - 1);
    bf16 q0 = (bf16)(acc0[v] + bias0);
    bf16 q1 = (bf16)(acc1[v] + bias1);
    bf16 q2 = (bf16)(acc2[v] + bias2);
    bf16 q3 = (bf16)(acc3[v] + bias3);
    if (mat == 2) {
      bf16* dst = vto + ((size_t)(b_ * NH + h) * HD) * SEQ + s;  // [d][s]
      dst[(size_t)( 0 + ll) * SEQ] = q0;
      dst[(size_t)(16 + ll) * SEQ] = q1;
      dst[(size_t)(32 + ll) * SEQ] = q2;
      dst[(size_t)(48 + ll) * SEQ] = q3;
    } else {
      bf16* dst = ((mat == 0) ? qo : ko) +
                  ((size_t)(b_ * NH + h) * SEQ + s) * HD;        // [s][d]
      dst[ 0 + ll] = q0;
      dst[16 + ll] = q1;
      dst[32 + ll] = q2;
      dst[48 + ll] = q3;
    }
  }
}

// ---------------------------------------------------------------------------
// Flash attention + relative_key_query bias. One wave per (b, h, 16-row tile).
// Per 32-column step: 2x(QK^T 2 WMMA + band bias 8 WMMA) + 4 PV WMMA.
// Band results and P are bounced through per-wave LDS regions (in-order DS).
// Next iteration's K/V/E lines are prefetched (global_prefetch_b8) to overlap
// L2->WGP latency with the current step's WMMAs.
// ---------------------------------------------------------------------------
__global__ void __launch_bounds__(128)
k_attn(const bf16* __restrict__ qb, const bf16* __restrict__ kb,
       const bf16* __restrict__ vt, const bf16* __restrict__ eb,
       const float* __restrict__ mask, float* __restrict__ out) {
  __shared__ float dqls[4][16][32];                 // Q @ Eslice band
  __shared__ float dkls[4][16][32];                 // K @ Eslice band
  __shared__ __align__(16) bf16 pls[4][16][32];     // probs re-layout for A-op

  const int lane = threadIdx.x & 31, wid = threadIdx.x >> 5;
  const int lh = lane >> 4, ll = lane & 15;
  const int LT = SEQ / 16;                          // 64 l-tiles
  int wg = blockIdx.x * 4 + wid;                    // 0 .. 3071
  int bh = wg / LT;
  int lt = wg - bh * LT;
  int b_ = bh / NH, h = bh - b_ * NH;
  int l0 = lt * 16;

  const bf16* Qh = qb + (size_t)bh * SEQ * HD;
  const bf16* Kh = kb + (size_t)bh * SEQ * HD;
  const bf16* Vh = vt + (size_t)bh * HD * SEQ;
  const float* mrow = mask + (size_t)b_ * SEQ;

  const bf16* qrow = Qh + (size_t)(l0 + ll) * HD;
  v16bf QA0 = ld_ab(qrow +      16 * lh, qrow +  8 + 16 * lh);   // d 0..31
  v16bf QA1 = ld_ab(qrow + 32 + 16 * lh, qrow + 40 + 16 * lh);   // d 32..63

  v8f O0 = {}, O1 = {}, O2 = {}, O3 = {};
  float mi[8], li[8];
#pragma unroll
  for (int v = 0; v < 8; ++v) { mi[v] = -1e30f; li[v] = 0.f; }

  for (int r0 = 0; r0 < SEQ; r0 += 32) {
    // prefetch next step's K rows (1 cacheline per row) and V^T segments
    if (r0 + 32 < SEQ) {
      __builtin_prefetch(Kh + (size_t)(r0 + 32 + lane) * HD, 0, 3);
      __builtin_prefetch(Vh + (size_t)(lane * 2) * SEQ + r0 + 32, 0, 3);
      __builtin_prefetch(Vh + (size_t)(lane * 2 + 1) * SEQ + r0 + 32, 0, 3);
      __builtin_prefetch(eb + (size_t)(l0 - r0 - 32 + 1008 + lane) * HD, 0, 3);
    }

    v8f S[2];
#pragma unroll
    for (int t = 0; t < 2; ++t) {
      int rt = r0 + 16 * t;
      const bf16* krow = Kh + (size_t)(rt + ll) * HD;

      // scores = Q K^T  (B-operand: col N = r-index, contiguous d per lane)
      v8f s = {};
      s = wmma_bf16(QA0, ld_b16(krow +      16 * lh), s);
      s = wmma_bf16(QA1, ld_b16(krow + 32 + 16 * lh), s);

      // diagonal band of dist_emb: j = base-15 .. base+16  (j = base+16 can be row 2047, past the table; never gathered)
      int base = l0 - rt + 1023;
      const bf16* e0 = eb + (size_t)(base - 15 + ll) * HD;     // c = ll
      const bf16* e1 = eb + (size_t)(base +  1 + ll) * HD;     // c = 16+ll
      v16bf EB00 = ld_b16(e0 +      16 * lh);                  // d 0..31
      v16bf EB01 = ld_b16(e0 + 32 + 16 * lh);                  // d 32..63
      v16bf EB10 = ld_b16(e1 +      16 * lh);
      v16bf EB11 = ld_b16(e1 + 32 + 16 * lh);

      v8f dq0 = {}, dq1 = {}, dk0 = {}, dk1 = {};
      dq0 = wmma_bf16(QA0, EB00, dq0); dq0 = wmma_bf16(QA1, EB01, dq0);
      dq1 = wmma_bf16(QA0, EB10, dq1); dq1 = wmma_bf16(QA1, EB11, dq1);

      v16bf KA0 = ld_ab(krow +      16 * lh, krow +  8 + 16 * lh);
      v16bf KA1 = ld_ab(krow + 32 + 16 * lh, krow + 40 + 16 * lh);
      dk0 = wmma_bf16(KA0, EB00, dk0); dk0 = wmma_bf16(KA1, EB01, dk0);
      dk1 = wmma_bf16(KA0, EB10, dk1); dk1 = wmma_bf16(KA1, EB11, dk1);

      // stage bands in this wave's LDS region
#pragma unroll
      for (int v = 0; v < 8; ++v) {
        int M = v + 8 * lh;
        dqls[wid][M][ll]      = dq0[v];
        dqls[wid][M][16 + ll] = dq1[v];
        dkls[wid][M][ll]      = dk0[v];
        dkls[wid][M][16 + ll] = dk1[v];
      }
      __builtin_amdgcn_wave_barrier();   // keep DS write->read ordering

      // gather: bias(M,N) = Dq[M, 15+M-N] + Dk[N, 15+M-N]
      float mv = mrow[rt + ll];
#pragma unroll
      for (int v = 0; v < 8; ++v) {
        int M = v + 8 * lh;
        int c = 15 + M - ll;
        float sv = s[v] + dqls[wid][M][c] + dkls[wid][ll][c];
        S[t][v] = sv * 0.125f + mv;      // 1/sqrt(64), additive mask
      }
    }

    // online softmax over the 32 new columns (row = M, cols across lane half)
#pragma unroll
    for (int v = 0; v < 8; ++v) {
      float loc = fmaxf(S[0][v], S[1][v]);
      loc = fmaxf(loc, __shfl_xor(loc, 1, 32));
      loc = fmaxf(loc, __shfl_xor(loc, 2, 32));
      loc = fmaxf(loc, __shfl_xor(loc, 4, 32));
      loc = fmaxf(loc, __shfl_xor(loc, 8, 32));
      float mnew = fmaxf(mi[v], loc);
      float corr = __expf(mi[v] - mnew);
      float p0 = __expf(S[0][v] - mnew);
      float p1 = __expf(S[1][v] - mnew);
      float rs = p0 + p1;
      rs += __shfl_xor(rs, 1, 32);
      rs += __shfl_xor(rs, 2, 32);
      rs += __shfl_xor(rs, 4, 32);
      rs += __shfl_xor(rs, 8, 32);
      li[v] = li[v] * corr + rs;
      mi[v] = mnew;
      O0[v] *= corr; O1[v] *= corr; O2[v] *= corr; O3[v] *= corr;
      int M = v + 8 * lh;
      pls[wid][M][ll]      = (bf16)p0;   // C-layout -> LDS for A-op re-layout
      pls[wid][M][16 + ll] = (bf16)p1;
    }
    __builtin_amdgcn_wave_barrier();

    // O += P(16x32) @ V(32x64): A from LDS, B from V^T (contiguous r per lane)
    v16bf PA = ld_ab(&pls[wid][ll][16 * lh], &pls[wid][ll][16 * lh + 8]);
    const bf16* vcol = Vh + (size_t)ll * SEQ + r0 + 16 * lh;
    O0 = wmma_bf16(PA, ld_b16(vcol + (size_t)(0 * 16) * SEQ), O0);
    O1 = wmma_bf16(PA, ld_b16(vcol + (size_t)(1 * 16) * SEQ), O1);
    O2 = wmma_bf16(PA, ld_b16(vcol + (size_t)(2 * 16) * SEQ), O2);
    O3 = wmma_bf16(PA, ld_b16(vcol + (size_t)(3 * 16) * SEQ), O3);
  }

  // normalize and scatter f32 output [b][s][h*64+d]
#pragma unroll
  for (int v = 0; v < 8; ++v) {
    float inv = 1.f / li[v];
    int M = v + 8 * lh;
    float* orow = out + ((size_t)(b_ * SEQ + l0 + M)) * HID + h * HD;
    orow[ 0 + ll] = O0[v] * inv;
    orow[16 + ll] = O1[v] * inv;
    orow[32 + ll] = O2[v] * inv;
    orow[48 + ll] = O3[v] * inv;
  }
}

// ---------------------------------------------------------------------------
extern "C" void kernel_launch(void* const* d_in, const int* in_sizes, int n_in,
                              void* d_out, int out_size, void* d_ws, size_t ws_size,
                              hipStream_t stream) {
  const float* hs   = (const float*)d_in[0];
  const float* mask = (const float*)d_in[1];
  const float* Wq   = (const float*)d_in[2];
  const float* bq   = (const float*)d_in[3];
  const float* Wk   = (const float*)d_in[4];
  const float* bk   = (const float*)d_in[5];
  const float* Wv   = (const float*)d_in[6];
  const float* bv   = (const float*)d_in[7];
  const float* de   = (const float*)d_in[8];

  char* ws = (char*)d_ws;
  bf16* hb  = (bf16*)(ws + OFF_HB);
  bf16* wtq = (bf16*)(ws + OFF_WTQ);
  bf16* wtk = (bf16*)(ws + OFF_WTK);
  bf16* wtv = (bf16*)(ws + OFF_WTV);
  bf16* ebf = (bf16*)(ws + OFF_EB);
  bf16* qbf = (bf16*)(ws + OFF_Q);
  bf16* kbf = (bf16*)(ws + OFF_K);
  bf16* vtb = (bf16*)(ws + OFF_VT);

  // bf16 conversions (one-shot, bandwidth-trivial)
  k_cvt<<<2048, 256, 0, stream>>>(hs, hb, NTOK * HID);
  k_cvt_wt<<<(HID * HID + 255) / 256, 256, 0, stream>>>(Wq, wtq);
  k_cvt_wt<<<(HID * HID + 255) / 256, 256, 0, stream>>>(Wk, wtk);
  k_cvt_wt<<<(HID * HID + 255) / 256, 256, 0, stream>>>(Wv, wtv);
  k_cvt<<<512, 256, 0, stream>>>(de, ebf, EROWS * HD);

  // QKV projection: 3 * 256 * 12 wave-tiles (16x64 each), 4 waves / block
  k_qkv<<<(3 * (NTOK / 16) * (HID / 64)) / 4, 128, 0, stream>>>(
      hb, wtq, wtk, wtv, bq, bk, bv, qbf, kbf, vtb);

  // attention: 4*12*64 waves, 4 per block
  k_attn<<<(BATCH * NH * (SEQ / 16)) / 4, 128, 0, stream>>>(
      qbf, kbf, vtb, ebf, mask, (float*)d_out);
}
